// NodeReplication_14757507629491
// MI455X (gfx1250) — compile-verified, matrix operand order checked by reading, not run on hardware
//
#include <hip/hip_runtime.h>
#include <hip/hip_bf16.h>
#include <stdint.h>

// ---------------------------------------------------------------------------
// NodeReplication on MI455X (gfx1250, wave32, WMMA)
//   N=4096 nodes, D=256, K_MAX=128, 2N=8192 candidate columns.
//   Heavy work: two 4096x8192x256 GEMMs -> bf16 v_wmma_f32_16x16x32_bf16.
//   theta (128MB) is never materialized: fused radix-threshold top-128
//   select per row, recomputing the score tile on the emission pass.
//   k-tiles are staged global->LDS with async B128 copies (ASYNCcnt) when
//   the toolchain exposes the gfx1250 async builtins; direct-global fallback
//   otherwise.
// ---------------------------------------------------------------------------

#define N_NODES 4096
#define DDIM    256
#define N2      8192      // 2*N_NODES candidate columns
#define KMAX    128
#define NE      131072    // edges
#define TEMP    1.0578255f  // (1e100)^(1/4096)

typedef __attribute__((ext_vector_type(16))) __bf16 v16bf;
typedef __attribute__((ext_vector_type(8)))  float  v8f;

#if defined(__AMDGCN__) && __has_builtin(__builtin_amdgcn_global_load_async_to_lds_b128) && \
    __has_builtin(__builtin_amdgcn_s_wait_asynccnt)
#define HAVE_ASYNC_LDS 1
#else
#define HAVE_ASYNC_LDS 0
#endif

__device__ __forceinline__ v8f wmma_bf16(const v16bf a, const v16bf b, v8f c) {
  return __builtin_amdgcn_wmma_f32_16x16x32_bf16(false, a, false, b, (short)0, c,
                                                 false, false);
}

__device__ __forceinline__ uint32_t hash_u32(uint32_t x) {
  x ^= x >> 16; x *= 0x7feb352dU; x ^= x >> 15; x *= 0x846ca68bU; x ^= x >> 16;
  return x;
}
__device__ __forceinline__ float hu(uint32_t a, uint32_t salt) {
  return (float)(hash_u32(a * 0x9E3779B9u ^ hash_u32(salt)) >> 8) * (1.0f / 16777216.0f);
}
__device__ __forceinline__ float sigmoidf_(float x) { return 1.f / (1.f + __expf(-x)); }
// monotonic f32 -> u32 key (descending float order == descending key order)
__device__ __forceinline__ uint32_t fkey(float v) {
  uint32_t b = __float_as_uint(v);
  return (b & 0x80000000u) ? ~b : (b | 0x80000000u);
}

#if HAVE_ASYNC_LDS
// Builtin signature (from hipcc diagnostic): param0 = AS1 pointer to GCC-style
// v4i32, param1 = AS3 pointer to v4i32, then two imm ints (offset, cpol).
typedef int v4i_gcc __attribute__((vector_size(16)));
typedef __attribute__((address_space(1))) v4i_gcc gv4i;
typedef __attribute__((address_space(3))) v4i_gcc lv4i;
// generic->AS casts via integer: AS(3) pointers are the low 32 bits of the
// generic shared address (LDS aperture maps addr[31:0] to the LDS offset).
__device__ __forceinline__ void async_copy_b128(const void* g, void* l) {
  __builtin_amdgcn_global_load_async_to_lds_b128(
      (gv4i*)(uintptr_t)g, (lv4i*)(uint32_t)(uintptr_t)l, 0, 0);
}
// Stage half h (128 K-elems) of a 16-column k-tile: 16 rows x 256B, 8 async
// B128 ops per wave (each op = 32 lanes x 16B = 2 rows).
__device__ __forceinline__ void stage_half(const __bf16* Kb, int jbase, int h,
                                           __bf16* buf, int lane) {
  const int ro = lane >> 4;        // row parity within the 2-row chunk
  const int co = (lane & 15) * 8;  // 16B per lane
#pragma unroll
  for (int i = 0; i < 8; ++i) {
    const __bf16* g = Kb + (size_t)(jbase + 2 * i + ro) * DDIM + h * 128 + co;
    __bf16* l = buf + (2 * i + ro) * 128 + co;
    async_copy_b128((const void*)g, (void*)l);
  }
}
#endif

// ---------------------------------------------------------------------------
// K1: replication scores, node mask, x_all (f32 out + bf16 mirror), src/dst=-1
// grid 512 x block 256 : one wave per node (wave32), 8 f32 per lane.
// ---------------------------------------------------------------------------
__global__ void nr_prep(const float* __restrict__ x,
                        const float* __restrict__ w_score,
                        const float* __restrict__ b_score,
                        float* __restrict__ out_xall,
                        __bf16* __restrict__ xall_bf,
                        uint32_t* __restrict__ mask,
                        int* __restrict__ out_src, int* __restrict__ out_dst) {
  const int tid = blockIdx.x * blockDim.x + threadIdx.x;  // 0..131071
#pragma unroll
  for (int t = 0; t < 4; ++t) {  // 131072*4 == N*KMAX slots each
    out_src[(size_t)tid * 4 + t] = -1;
    out_dst[(size_t)tid * 4 + t] = -1;
  }
  const int node = tid >> 5;
  const int lane = tid & 31;
  const float* xr = x + (size_t)node * DDIM + lane * 8;
  const float* wr = w_score + lane * 8;
  const float4 a0 = *(const float4*)xr,      a1 = *(const float4*)(xr + 4);
  const float4 b0 = *(const float4*)wr,      b1 = *(const float4*)(wr + 4);
  float s = a0.x*b0.x + a0.y*b0.y + a0.z*b0.z + a0.w*b0.w
          + a1.x*b1.x + a1.y*b1.y + a1.z*b1.z + a1.w*b1.w;
  for (int o = 16; o; o >>= 1) s += __shfl_xor(s, o);
  int m = 0;
  if (lane == 0) {
    const float logit = sigmoidf_((s + b_score[0]) * TEMP);
    m = (hu((uint32_t)node, 0xA341316Cu) < logit) ? 1 : 0;
    mask[node] = (uint32_t)m;
  }
  m = __shfl(m, 0);
  const float fm = m ? 1.f : 0.f;
  float*  o0 = out_xall + (size_t)node * DDIM + lane * 8;
  float*  o1 = out_xall + (size_t)(node + N_NODES) * DDIM + lane * 8;
  __bf16* p0 = xall_bf + (size_t)node * DDIM + lane * 8;
  __bf16* p1 = xall_bf + (size_t)(node + N_NODES) * DDIM + lane * 8;
  const float v[8] = {a0.x, a0.y, a0.z, a0.w, a1.x, a1.y, a1.z, a1.w};
#pragma unroll
  for (int t = 0; t < 8; ++t) {
    o0[t] = v[t];
    o1[t] = v[t] * fm;
    p0[t] = (__bf16)v[t];
    p1[t] = (__bf16)(v[t] * fm);
  }
}

// ---------------------------------------------------------------------------
// K2: transpose + convert projection weights to bf16 [n][k] so WMMA B-tiles
// are contiguous per lane. grid (256, 2) x 256.
// ---------------------------------------------------------------------------
__global__ void nr_convw(const float* __restrict__ wq, const float* __restrict__ wk,
                         __bf16* __restrict__ wqt, __bf16* __restrict__ wkt) {
  const int n = blockIdx.x, k = threadIdx.x;
  if (blockIdx.y == 0) wqt[(size_t)n * DDIM + k] = (__bf16)wq[(size_t)k * DDIM + n];
  else                 wkt[(size_t)n * DDIM + k] = (__bf16)wk[(size_t)k * DDIM + n];
}

// ---------------------------------------------------------------------------
// K3: projection GEMM  C[M,256] = A[M,256] @ W + bias ; outputs bf16, with
// optional per-column diagonal scalings (w_kai / w_theta folded in).
// One wave per 16x16 C tile; K loop 256/32 = 8 WMMAs.
// ---------------------------------------------------------------------------
__global__ void nr_gemm_wmma(const __bf16* __restrict__ A,
                             const __bf16* __restrict__ Wt,
                             const float* __restrict__ bias,
                             __bf16* __restrict__ out0, const float* __restrict__ scale0,
                             __bf16* __restrict__ out1, const float* __restrict__ scale1) {
  const int mbase = blockIdx.x * 16, nbase = blockIdx.y * 16;
  const int lane = threadIdx.x;
  const int l = lane & 15, half = lane >> 4;
  v8f acc = {};
  const __bf16* ap = A  + (size_t)(mbase + l) * DDIM + half * 16;
  const __bf16* bp = Wt + (size_t)(nbase + l) * DDIM + half * 16;
#pragma unroll
  for (int kk = 0; kk < DDIM; kk += 32) {
    const v16bf a = *(const v16bf*)(ap + kk);
    const v16bf b = *(const v16bf*)(bp + kk);
    acc = wmma_bf16(a, b, acc);
  }
  const int col = nbase + l;
  const float bv = bias[col];
  const float s0 = scale0 ? scale0[col] : 1.f;
#pragma unroll
  for (int r = 0; r < 8; ++r) {
    const int row = mbase + r + half * 8;  // C layout: vgpr r, lanes16-31 -> M+8
    const float v = acc[r] + bv;
    out0[(size_t)row * DDIM + col] = (__bf16)(v * s0);
    if (out1) out1[(size_t)row * DDIM + col] = (__bf16)(v * scale1[col]);
  }
}

// ---------------------------------------------------------------------------
// K4: fused scoring + top-128 select. One block = 4 waves handles a 16-row
// strip of q against all 8192 k-columns.
//  Pass A: WMMA s_kai & s_theta tiles -> LDS radix histogram + kai relu-sum.
//  Threshold: per-row 128th-largest key bin (+ tie quota).
//  Pass B: recompute s_theta tiles, emit (src,dst) for selected slots with
//  bernoulli(theta) * mask * (slot < k_eff) validity.
// k-tiles stream through LDS via async B128 copies (split-K double buffer,
// 8 ops in flight per half, s_wait_asynccnt(8) before each consume).
// ---------------------------------------------------------------------------
__global__ void __launch_bounds__(128) nr_score_select(
    const __bf16* __restrict__ Akai, const __bf16* __restrict__ Ath,
    const __bf16* __restrict__ Kb,   const uint32_t* __restrict__ mask,
    const float* __restrict__ b_kai_p, const float* __restrict__ b_theta_p,
    int* __restrict__ out_src, int* __restrict__ out_dst) {
  __shared__ uint32_t hist[16][256];
  __shared__ float    lds_kai[16];
  __shared__ int      thresh[16], tiequota[16], keff_s[16], maskrow[16];
  __shared__ uint32_t tiecnt[16], cnt[16];
#if HAVE_ASYNC_LDS
  __shared__ __bf16 kstage[4][2][16 * 128];  // [wave][K-half][16 cols x 128 K] : 32 KB
#endif

  const int mbase = blockIdx.x * 16;
  const int tid = threadIdx.x;
  const int wave = tid >> 5, lane = tid & 31;
  const int l = lane & 15, half = lane >> 4;
  const float b_kai = b_kai_p[0], b_theta = b_theta_p[0];
  const int NT = N2 / 16;

  { uint32_t* h = &hist[0][0];
    for (int i = tid; i < 16 * 256; i += 128) h[i] = 0; }
  if (tid < 16) { lds_kai[tid] = 0.f; tiecnt[tid] = 0; cnt[tid] = 0; }
  __syncthreads();

  // A-fragments (both scaled-q matrices) are invariant across column tiles.
  v16bf afk[8], aft[8];
#pragma unroll
  for (int ks = 0; ks < 8; ++ks) {
    const size_t off = (size_t)(mbase + l) * DDIM + ks * 32 + half * 16;
    afk[ks] = *(const v16bf*)(Akai + off);
    aft[ks] = *(const v16bf*)(Ath + off);
  }

#if HAVE_ASYNC_LDS
  __bf16* myst0 = &kstage[wave][0][0];
  __bf16* myst1 = &kstage[wave][1][0];
#endif

  // ---- Pass A: histogram of theta keys + kai relu-sum ----
#if HAVE_ASYNC_LDS
  stage_half(Kb, wave * 16, 0, myst0, lane);
#endif
  for (int jt = wave; jt < NT; jt += 4) {
    const int jbase = jt * 16;
    v8f ak = {}, at = {};
#if HAVE_ASYNC_LDS
    const int jnext = (jt + 4 < NT) ? (jt + 4) * 16 : jbase;
    stage_half(Kb, jbase, 1, myst1, lane);
    __builtin_amdgcn_s_wait_asynccnt(8);  // half0(jt) resident
#pragma unroll
    for (int ks = 0; ks < 4; ++ks) {
      const v16bf b = *(const v16bf*)(myst0 + l * 128 + ks * 32 + half * 16);
      ak = wmma_bf16(afk[ks], b, ak);
      at = wmma_bf16(aft[ks], b, at);
    }
    stage_half(Kb, jnext, 0, myst0, lane);  // prefetch next tile's half0
    __builtin_amdgcn_s_wait_asynccnt(8);    // half1(jt) resident
#pragma unroll
    for (int ks = 4; ks < 8; ++ks) {
      const v16bf b = *(const v16bf*)(myst1 + l * 128 + (ks - 4) * 32 + half * 16);
      ak = wmma_bf16(afk[ks], b, ak);
      at = wmma_bf16(aft[ks], b, at);
    }
#else
    const __bf16* kp = Kb + (size_t)(jbase + l) * DDIM + half * 16;
    __builtin_prefetch(Kb + (size_t)(jbase + 64 + l) * DDIM, 0, 1);
#pragma unroll
    for (int ks = 0; ks < 8; ++ks) {
      const v16bf b = *(const v16bf*)(kp + ks * 32);
      ak = wmma_bf16(afk[ks], b, ak);
      at = wmma_bf16(aft[ks], b, at);
    }
#endif
#pragma unroll
    for (int r = 0; r < 8; ++r) {
      const int row = r + half * 8;
      atomicAdd(&lds_kai[row], fmaxf(ak[r] + b_kai, 0.f));
      const uint32_t k8 = fkey(at[r] + b_theta) >> 24;
      atomicAdd(&hist[row][k8], 1u);
    }
  }
#if HAVE_ASYNC_LDS
  __builtin_amdgcn_s_wait_asynccnt(0);
#endif
  __syncthreads();

  if (tid < 16) {
    int cum = 0, bin = 255, quota = KMAX;
    for (; bin >= 0; --bin) {
      const int c = (int)hist[tid][bin];
      if (cum + c >= KMAX) { quota = KMAX - cum; break; }
      cum += c;
    }
    if (bin < 0) { bin = 0; quota = KMAX; }
    thresh[tid] = bin;
    tiequota[tid] = quota;
    const int gi = mbase + tid;
    maskrow[tid] = (int)mask[gi];
    // k_samp ~ Poisson(kai) via normal approximation (kai is large here)
    const float lam = lds_kai[tid];
    const float u1 = hu((uint32_t)gi, 0x68E31DA4u) + 1e-7f;
    const float u2 = hu((uint32_t)gi, 0x1B56C4E9u);
    const float z = sqrtf(fmaxf(-2.f * __logf(u1), 0.f)) * __cosf(6.2831853f * u2);
    const int ks = (int)fmaxf(0.f, lam + sqrtf(fmaxf(lam, 0.f)) * z + 0.5f);
    keff_s[tid] = ks < KMAX ? ks : KMAX;
  }
  __syncthreads();

  // ---- Pass B: recompute theta tiles, emit selected slots ----
#if HAVE_ASYNC_LDS
  stage_half(Kb, wave * 16, 0, myst0, lane);
#endif
  for (int jt = wave; jt < NT; jt += 4) {
    const int jbase = jt * 16;
    v8f at = {};
#if HAVE_ASYNC_LDS
    const int jnext = (jt + 4 < NT) ? (jt + 4) * 16 : jbase;
    stage_half(Kb, jbase, 1, myst1, lane);
    __builtin_amdgcn_s_wait_asynccnt(8);
#pragma unroll
    for (int ks = 0; ks < 4; ++ks) {
      const v16bf b = *(const v16bf*)(myst0 + l * 128 + ks * 32 + half * 16);
      at = wmma_bf16(aft[ks], b, at);
    }
    stage_half(Kb, jnext, 0, myst0, lane);
    __builtin_amdgcn_s_wait_asynccnt(8);
#pragma unroll
    for (int ks = 4; ks < 8; ++ks) {
      const v16bf b = *(const v16bf*)(myst1 + l * 128 + (ks - 4) * 32 + half * 16);
      at = wmma_bf16(aft[ks], b, at);
    }
#else
    const __bf16* kp = Kb + (size_t)(jbase + l) * DDIM + half * 16;
#pragma unroll
    for (int ks = 0; ks < 8; ++ks) {
      const v16bf b = *(const v16bf*)(kp + ks * 32);
      at = wmma_bf16(aft[ks], b, at);
    }
#endif
#pragma unroll
    for (int r = 0; r < 8; ++r) {
      const int row = r + half * 8;
      const float sv = at[r] + b_theta;
      const int k8 = (int)(fkey(sv) >> 24);
      const int th = thresh[row];
      bool elig = (k8 > th);
      if (!elig && k8 == th)
        elig = ((int)atomicAdd(&tiecnt[row], 1u) < tiequota[row]);
      if (elig) {
        const int slot = (int)atomicAdd(&cnt[row], 1u);
        if (slot < KMAX) {
          const int gi = mbase + row;
          const int col = jbase + l;
          const float theta = sigmoidf_(sv * TEMP);
          const bool bern = hu((uint32_t)(gi * N2 + col), 0xB5297A4Du) < theta;
          const bool valid = bern && maskrow[row] && (slot < keff_s[row]);
          out_src[(size_t)gi * KMAX + slot] = valid ? (gi + N_NODES) : -1;
          out_dst[(size_t)gi * KMAX + slot] = valid ? col : -1;
        }
      }
    }
  }
}

// ---------------------------------------------------------------------------
extern "C" void kernel_launch(void* const* d_in, const int* in_sizes, int n_in,
                              void* d_out, int out_size, void* d_ws, size_t ws_size,
                              hipStream_t stream) {
  const float* x       = (const float*)d_in[0];
  const float* w_score = (const float*)d_in[1];
  const float* b_score = (const float*)d_in[2];
  const float* w_q     = (const float*)d_in[3];
  const float* b_q     = (const float*)d_in[4];
  const float* w_k     = (const float*)d_in[5];
  const float* b_k     = (const float*)d_in[6];
  const float* w_kai   = (const float*)d_in[7];
  const float* b_kai   = (const float*)d_in[8];
  const float* w_theta = (const float*)d_in[9];
  const float* b_theta = (const float*)d_in[10];
  const void*  edge    = d_in[11];  // int64 [2, E]

  // workspace layout (~12.5 MB)
  char* wsb = (char*)d_ws;
  size_t o = 0;
  __bf16*   xall_bf = (__bf16*)(wsb + o);   o += (size_t)N2 * DDIM * 2;
  uint32_t* mask    = (uint32_t*)(wsb + o); o += (size_t)N_NODES * 4;
  __bf16*   wqt     = (__bf16*)(wsb + o);   o += (size_t)DDIM * DDIM * 2;
  __bf16*   wkt     = (__bf16*)(wsb + o);   o += (size_t)DDIM * DDIM * 2;
  __bf16*   kbf     = (__bf16*)(wsb + o);   o += (size_t)N2 * DDIM * 2;
  __bf16*   akai    = (__bf16*)(wsb + o);   o += (size_t)N_NODES * DDIM * 2;
  __bf16*   ath     = (__bf16*)(wsb + o);   o += (size_t)N_NODES * DDIM * 2;

  // output layout: x_all f32 | edge_index i64 | src i32 | dst i32
  char* outb = (char*)d_out;
  float* out_xall = (float*)outb;
  size_t oo = (size_t)2 * N_NODES * DDIM * 4;
  void* out_edge = (void*)(outb + oo);      oo += (size_t)2 * NE * 8;
  int* out_src = (int*)(outb + oo);         oo += (size_t)N_NODES * KMAX * 4;
  int* out_dst = (int*)(outb + oo);

  nr_prep<<<512, 256, 0, stream>>>(x, w_score, b_score, out_xall, xall_bf, mask,
                                   out_src, out_dst);
  (void)hipMemcpyAsync(out_edge, edge, (size_t)2 * NE * 8, hipMemcpyDeviceToDevice,
                       stream);
  nr_convw<<<dim3(256, 2), 256, 0, stream>>>(w_q, w_k, wqt, wkt);
  // k = x_all @ w_k + b_k  (8192 rows)
  nr_gemm_wmma<<<dim3(N2 / 16, 16), 32, 0, stream>>>(xall_bf, wkt, b_k, kbf, nullptr,
                                                     nullptr, nullptr);
  // q = new_x @ w_q + b_q, fused with w_kai / w_theta diagonal scalings
  nr_gemm_wmma<<<dim3(N_NODES / 16, 16), 32, 0, stream>>>(
      xall_bf + (size_t)N_NODES * DDIM, wqt, b_q, akai, w_kai, ath, w_theta);
  nr_score_select<<<N_NODES / 16, 128, 0, stream>>>(akai, ath, kbf, mask, b_kai,
                                                    b_theta, out_src, out_dst);
}
